// PairMLPCond_79869211836485
// MI455X (gfx1250) — matrix operand order repaired and checked by reading, not compiled, not run on hardware
//
#include <hip/hip_runtime.h>
#include <hip/hip_bf16.h>

typedef __attribute__((ext_vector_type(16))) _Float16 v16h;
typedef __attribute__((ext_vector_type(8)))  _Float16 v8h;
typedef __attribute__((ext_vector_type(8)))  float    v8f;

#define ZDIM 64
#define HID  128
#define CDIM 4

union U16h { v16h v; v8h h[2]; };

// CDNA5 async global->LDS copy (16B per lane), tracked by ASYNCcnt.
__device__ __forceinline__ void async_copy16(const void* gsrc, unsigned lds_off) {
    asm volatile("global_load_async_to_lds_b128 %0, %1, off"
                 :: "v"(lds_off), "v"((unsigned long long)(uintptr_t)gsrc)
                 : "memory");
}
__device__ __forceinline__ void wait_async0() {
    asm volatile("s_wait_asynccnt 0x0" ::: "memory");
}

// ---------------- elementwise f32 -> f16 conversion of the z table ----------
__global__ void k_cvt_z(const float* __restrict__ z, _Float16* __restrict__ zh, int n) {
    int i = blockIdx.x * blockDim.x + threadIdx.x;
    if (i < n) zh[i] = (_Float16)z[i];
}

// ---------------- weight prep: transpose to B-layout, fold c into bias ------
__global__ void k_prep(const float* __restrict__ W1, const float* __restrict__ b1,
                       const float* __restrict__ c,  const float* __restrict__ W2,
                       _Float16* __restrict__ W1aT, _Float16* __restrict__ W1bT,
                       _Float16* __restrict__ W2T,
                       float* __restrict__ b1p, float* __restrict__ sums,
                       float* __restrict__ sumsq) {
    int n = threadIdx.x;
    if (n < HID) {
        // b1' = b1 + c @ W1[128:132, :]   (c is constant over edges)
        float acc = b1[n];
        #pragma unroll
        for (int i = 0; i < CDIM; ++i) acc += c[i] * W1[(2 * ZDIM + i) * HID + n];
        b1p[n]  = acc;
        sums[n] = 0.f;   // zero stats every call (deterministic)
        sumsq[n] = 0.f;
        for (int k = 0; k < ZDIM; ++k) {
            W1aT[n * ZDIM + k] = (_Float16)W1[k * HID + n];            // W1[0:64,:]^T
            W1bT[n * ZDIM + k] = (_Float16)W1[(ZDIM + k) * HID + n];   // W1[64:128,:]^T
        }
        for (int k = 0; k < HID; ++k) W2T[n * HID + k] = (_Float16)W2[k * HID + n];
    }
}

// ---------------- BN statistics -> fused scale/shift -------------------------
__global__ void k_stats(const float* __restrict__ sums, const float* __restrict__ sumsq,
                        const float* __restrict__ gamma, const float* __restrict__ beta,
                        float invE, float* __restrict__ scale, float* __restrict__ shift) {
    int f = threadIdx.x;
    if (f < HID) {
        float m = sums[f] * invE;
        float v = sumsq[f] * invE - m * m;
        float r = rsqrtf(v + 1e-5f);
        float sc = gamma[f] * r;
        scale[f] = sc;
        shift[f] = beta[f] - m * sc;
    }
}

// ---------------- main fused kernel: PHASE 0 = stats pass, PHASE 1 = output --
template <int PHASE>
__global__ void __launch_bounds__(128, 2)
k_main(const _Float16* __restrict__ zh, const long long* __restrict__ ei,
       int E, int nTiles,
       const _Float16* __restrict__ gW1a, const _Float16* __restrict__ gW1b,
       const _Float16* __restrict__ gW2T,
       const float* __restrict__ b1p, const float* __restrict__ b2,
       const float* __restrict__ scale, const float* __restrict__ shift,
       const float* __restrict__ W3, const float* __restrict__ b3,
       float* __restrict__ sums, float* __restrict__ sumsq,
       float* __restrict__ out) {
    // Pass A: W1a|W1b in LDS (32 KB).  Pass B: W2 in LDS (32 KB) + 4x4 KB h-staging.
    __shared__ __align__(32) _Float16 sW1[(PHASE == 0) ? 2 * HID * ZDIM : 8];
    __shared__ __align__(32) _Float16 sW2[(PHASE == 1) ? HID * HID : 8];
    __shared__ __align__(32) _Float16 sH [(PHASE == 1) ? 4 * 16 * HID : 8];

    const int tid = threadIdx.x;
    // ---- CDNA5 async global->LDS weight staging (bypasses VGPRs) ----
    if (PHASE == 0) {
        const unsigned base0 = (unsigned)(uintptr_t)&sW1[0];
        const unsigned base1 = base0 + HID * ZDIM * 2;
        for (int i = tid; i < HID * ZDIM / 8; i += blockDim.x) {
            async_copy16((const uint4*)gW1a + i, base0 + i * 16);
            async_copy16((const uint4*)gW1b + i, base1 + i * 16);
        }
    } else {
        const unsigned base2 = (unsigned)(uintptr_t)&sW2[0];
        for (int i = tid; i < HID * HID / 8; i += blockDim.x)
            async_copy16((const uint4*)gW2T + i, base2 + i * 16);
    }
    wait_async0();
    __syncthreads();

    const int lane = tid & 31;
    const int wid  = tid >> 5;
    const int col  = lane & 15;   // WMMA B/C/D column, A row
    const int hi   = lane >> 4;   // K-half select

    float b1l[8], scl[8], shl[8], b2l[8], w3l[8];
    #pragma unroll
    for (int t = 0; t < 8; ++t) {
        b1l[t] = b1p[t * 16 + col];
        scl[t] = shl[t] = b2l[t] = w3l[t] = 0.f;
        if (PHASE == 1) {
            scl[t] = scale[t * 16 + col];
            shl[t] = shift[t * 16 + col];
            b2l[t] = b2[t * 16 + col];
            w3l[t] = W3[t * 16 + col];
        }
    }
    const float b3v = (PHASE == 1) ? b3[0] : 0.f;

    float sumA[8] = {0, 0, 0, 0, 0, 0, 0, 0};
    float sumQ[8] = {0, 0, 0, 0, 0, 0, 0, 0};

    const int nWaves = gridDim.x * (blockDim.x >> 5);
    const int wave0  = blockIdx.x * (blockDim.x >> 5) + wid;
    _Float16* myH = (PHASE == 1) ? &sH[wid * 16 * HID] : nullptr;

    // per-lane A-fragment gather for one 16-edge tile (ISA 16-bit A layout)
    auto loadFrags = [&](int t, v16h* AS, v16h* AD) {
        const int eIdx = t * 16 + col;
        const long long si = (eIdx < E) ? ei[eIdx] : 0;
        const long long di = (eIdx < E) ? ei[E + eIdx] : 0;
        const _Float16* zs = zh + si * ZDIM;
        const _Float16* zd = zh + di * ZDIM;
        #pragma unroll
        for (int kt = 0; kt < 2; ++kt) {
            const int o = kt * 32 + hi * 8;
            U16h u;
            u.h[0] = *(const v8h*)(zs + o); u.h[1] = *(const v8h*)(zs + o + 16); AS[kt] = u.v;
            u.h[0] = *(const v8h*)(zd + o); u.h[1] = *(const v8h*)(zd + o + 16); AD[kt] = u.v;
        }
    };

    v16h aS[2], aD[2];
    if (wave0 < nTiles) loadFrags(wave0, aS, aD);

    for (int tile = wave0; tile < nTiles; tile += nWaves) {
        const int base = tile * 16;
        const int nextTile = tile + nWaves;
        __builtin_prefetch(ei + base + nWaves * 16, 0, 0);

        // ---- software pipeline: issue next tile's gathers before computing ----
        v16h nS[2], nD[2];
        if (nextTile < nTiles) loadFrags(nextTile, nS, nD);

        // ---- layer 1: h1 = relu([z_src|z_dst] @ W1[:128] + b1') ----
        v8f acc[8];
        #pragma unroll
        for (int nt = 0; nt < 8; ++nt) {
            v8f a = {};
            const _Float16* w1aP = (PHASE == 0) ? &sW1[(nt * 16 + col) * ZDIM]
                                                : &gW1a[(nt * 16 + col) * ZDIM];
            const _Float16* w1bP = (PHASE == 0) ? &sW1[HID * ZDIM + (nt * 16 + col) * ZDIM]
                                                : &gW1b[(nt * 16 + col) * ZDIM];
            #pragma unroll
            for (int kt = 0; kt < 2; ++kt) {
                U16h ub;
                ub.h[0] = *(const v8h*)(w1aP + kt * 32 + hi * 8); ub.h[1] = *(const v8h*)(w1aP + kt * 32 + hi * 8 + 16);
                v16h bA = ub.v;
                a = __builtin_amdgcn_wmma_f32_16x16x32_f16(false, aS[kt], false, bA,
                                                           (short)0, a, false, false);
                ub.h[0] = *(const v8h*)(w1bP + kt * 32 + hi * 8); ub.h[1] = *(const v8h*)(w1bP + kt * 32 + hi * 8 + 16);
                v16h bB = ub.v;
                a = __builtin_amdgcn_wmma_f32_16x16x32_f16(false, aD[kt], false, bB,
                                                           (short)0, a, false, false);
            }
            #pragma unroll
            for (int r = 0; r < 8; ++r) a[r] = fmaxf(a[r] + b1l[nt], 0.f);
            acc[nt] = a;
        }

        if (PHASE == 0) {
            // register-accumulated per-feature stats (flushed once at the end)
            #pragma unroll
            for (int nt = 0; nt < 8; ++nt) {
                float s = 0.f, q = 0.f;
                #pragma unroll
                for (int r = 0; r < 8; ++r) { float x = acc[nt][r]; s += x; q += x * x; }
                sumA[nt] += s; sumQ[nt] += q;
            }
        } else {
            // ---- BatchNorm, then transpose C/D-layout -> A-layout via LDS ----
            #pragma unroll
            for (int nt = 0; nt < 8; ++nt) {
                #pragma unroll
                for (int r = 0; r < 8; ++r) {
                    float x = acc[nt][r] * scl[nt] + shl[nt];
                    myH[(hi * 8 + r) * HID + nt * 16 + col] = (_Float16)x; // M = hi*8+r
                }
            }
            asm volatile("s_wait_dscnt 0x0" ::: "memory");

            v16h a2[4];
            #pragma unroll
            for (int kt = 0; kt < 4; ++kt) {
                const int o = kt * 32 + hi * 8;
                U16h u;
                u.h[0] = *(const v8h*)&myH[col * HID + o];
                u.h[1] = *(const v8h*)&myH[col * HID + o + 16];
                a2[kt] = u.v;
            }

            // ---- layer 2: h2 = relu(h_bn @ W2 + b2) ----
            v8f acc2[8];
            #pragma unroll
            for (int nt = 0; nt < 8; ++nt) {
                v8f a = {};
                #pragma unroll
                for (int kt = 0; kt < 4; ++kt) {
                    U16h ub;
                    ub.h[0] = *(const v8h*)&sW2[(nt * 16 + col) * HID + kt * 32 + hi * 8];
                    ub.h[1] = *(const v8h*)&sW2[(nt * 16 + col) * HID + kt * 32 + hi * 8 + 16];
                    v16h b = ub.v;
                    a = __builtin_amdgcn_wmma_f32_16x16x32_f16(false, a2[kt], false, b,
                                                               (short)0, a, false, false);
                }
                #pragma unroll
                for (int r = 0; r < 8; ++r) a[r] = fmaxf(a[r] + b2l[nt], 0.f);
                acc2[nt] = a;
            }

            // ---- layer 3: logits = h2 @ W3 + b3 (wave32 butterfly reduce) ----
            #pragma unroll
            for (int r = 0; r < 8; ++r) {
                float p = 0.f;
                #pragma unroll
                for (int nt = 0; nt < 8; ++nt) p += acc2[nt][r] * w3l[nt];
                p += __shfl_xor(p, 1, 32);
                p += __shfl_xor(p, 2, 32);
                p += __shfl_xor(p, 4, 32);
                p += __shfl_xor(p, 8, 32);
                const int e = base + hi * 8 + r;
                if (col == 0 && e < E) out[e] = p + b3v;
            }
        }

        // rotate double buffer
        aS[0] = nS[0]; aS[1] = nS[1]; aD[0] = nD[0]; aD[1] = nD[1];
    }

    if (PHASE == 0) {
        #pragma unroll
        for (int nt = 0; nt < 8; ++nt) {
            atomicAdd(&sums[nt * 16 + col],  sumA[nt]);
            atomicAdd(&sumsq[nt * 16 + col], sumQ[nt]);
        }
    }
}

extern "C" void kernel_launch(void* const* d_in, const int* in_sizes, int n_in,
                              void* d_out, int out_size, void* d_ws, size_t ws_size,
                              hipStream_t stream) {
    (void)n_in; (void)out_size; (void)ws_size;
    const float*     z     = (const float*)d_in[0];
    const long long* ei    = (const long long*)d_in[1];
    const float*     c     = (const float*)d_in[2];
    const float*     W1    = (const float*)d_in[3];
    const float*     b1    = (const float*)d_in[4];
    const float*     gamma = (const float*)d_in[5];
    const float*     beta  = (const float*)d_in[6];
    const float*     W2    = (const float*)d_in[7];
    const float*     b2    = (const float*)d_in[8];
    const float*     W3    = (const float*)d_in[9];
    const float*     b3    = (const float*)d_in[10];
    float* out = (float*)d_out;

    const int nZ = in_sizes[0];        // N_NODES * 64
    const int E  = in_sizes[1] / 2;    // edges
    const int nTiles = (E + 15) / 16;

    char*  ws  = (char*)d_ws;
    size_t off = 0;
    auto alloc = [&](size_t bytes) { size_t o = off; off = (off + bytes + 255) & ~(size_t)255; return o; };

    _Float16* zh    = (_Float16*)(ws + alloc((size_t)nZ * 2));
    _Float16* W1aT  = (_Float16*)(ws + alloc((size_t)HID * ZDIM * 2));
    _Float16* W1bT  = (_Float16*)(ws + alloc((size_t)HID * ZDIM * 2));
    _Float16* W2T   = (_Float16*)(ws + alloc((size_t)HID * HID * 2));
    float*    b1p   = (float*)(ws + alloc(HID * 4));
    float*    sums  = (float*)(ws + alloc(HID * 4));
    float*    sumsq = (float*)(ws + alloc(HID * 4));
    float*    scl   = (float*)(ws + alloc(HID * 4));
    float*    shf   = (float*)(ws + alloc(HID * 4));

    k_cvt_z<<<(nZ + 255) / 256, 256, 0, stream>>>(z, zh, nZ);
    k_prep<<<1, 128, 0, stream>>>(W1, b1, c, W2, W1aT, W1bT, W2T, b1p, sums, sumsq);

    const int blocks = 512;  // 2048 waves -> ~49 tiles/wave; stats stay in registers
    k_main<0><<<blocks, 128, 0, stream>>>(zh, ei, E, nTiles, W1aT, W1bT, W2T,
                                          b1p, b2, scl, shf, W3, b3, sums, sumsq, out);
    k_stats<<<1, 128, 0, stream>>>(sums, sumsq, gamma, beta, 1.0f / (float)E, scl, shf);
    k_main<1><<<blocks, 128, 0, stream>>>(zh, ei, E, nTiles, W1aT, W1bT, W2T,
                                          b1p, b2, scl, shf, W3, b3, sums, sumsq, out);
}
